// CifMiddleware_6004364280186
// MI455X (gfx1250) — matrix operand order repaired and checked by reading, not compiled, not run on hardware
//
#include <hip/hip_runtime.h>
#include <hip/hip_bf16.h>
#include <math.h>

// ---------------- problem constants (from reference) ----------------
constexpr int Bn = 32, Tn = 2048, Cn = 512, Kc = 5;
constexpr int Tpad = Tn + 4;                 // 2-row halo each side per batch
constexpr int Mtot = Bn * Tn;                // 65536 GEMM rows
constexpr float THRESH = 1.0f;

typedef __attribute__((ext_vector_type(16))) __bf16 bf16x16;
typedef __attribute__((ext_vector_type(8)))  __bf16 bf16x8;
typedef __attribute__((ext_vector_type(8)))  float  f32x8;

union AFrag { bf16x16 v; bf16x8 h[2]; };
struct Frags { AFrag a0, a1, bb[4]; };       // 32x32 A slab + 4 B tiles (32x16 each)

__device__ __forceinline__ unsigned short f2bf(float f) {
    unsigned u = __float_as_uint(f);
    unsigned r = u + 0x7FFFu + ((u >> 16) & 1u);   // round-to-nearest-even
    return (unsigned short)(r >> 16);
}
__device__ __forceinline__ float bf2f(unsigned short h) {
    return __uint_as_float(((unsigned)h) << 16);
}
__device__ __forceinline__ f32x8 wmma_bf16(bf16x16 a, bf16x16 b, f32x8 c) {
    return __builtin_amdgcn_wmma_f32_16x16x32_bf16(
        /*neg_a=*/false, a, /*neg_b=*/false, b,
        /*c_mod=*/(short)0, c, /*reuse_a=*/false, /*reuse_b=*/false);
}

// ---------------- kernel 0: zero d_out (+wsum) ----------------
__global__ void k_init(float* out, long long n_out, float* wsum) {
    long long i = (long long)blockIdx.x * blockDim.x + threadIdx.x;
    if (i < n_out) out[i] = 0.0f;
    if (i < Bn)    wsum[i] = 0.0f;
}

// ---------------- kernel 1: x -> bf16 with per-batch 2-row halo ----------------
__global__ void k_prep_x(const float* __restrict__ x, unsigned short* __restrict__ xpad) {
    long long i = (long long)blockIdx.x * blockDim.x + threadIdx.x;  // over B*Tpad*C
    long long total = (long long)Bn * Tpad * Cn;
    if (i >= total) return;
    int c  = (int)(i % Cn);
    long long p = i / Cn;
    int b  = (int)(p / Tpad);
    int pr = (int)(p % Tpad);
    if (pr < 2 || pr >= Tn + 2) { xpad[i] = 0; return; }
    xpad[i] = f2bf(x[((long long)b * Tn + (pr - 2)) * Cn + c]);
}

// ---------------- kernel 2: transpose weights to bf16 ----------------
// convWT layout [k][o][i] = conv_w[o][i][k] ; denseWT layout [d][c] = dense_w[c][d]
__global__ void k_prep_w(const float* __restrict__ conv_w, const float* __restrict__ dense_w,
                         unsigned short* __restrict__ convWT, unsigned short* __restrict__ denseWT) {
    int i = blockIdx.x * blockDim.x + threadIdx.x;
    const int n1 = Kc * Cn * Cn;
    if (i < n1) {
        int k = i / (Cn * Cn);
        int r = i % (Cn * Cn);
        int o = r / Cn, ii = r % Cn;
        convWT[i] = f2bf(conv_w[((long long)o * Cn + ii) * Kc + k]);
    } else if (i < n1 + Cn * Cn) {
        int j = i - n1;
        int d = j / Cn, c = j % Cn;
        denseWT[j] = f2bf(dense_w[(long long)c * Cn + d]);
    }
}

// ---------------- kernel 3: conv1d as 5 accumulated bf16 WMMA GEMMs ----------------
// wave tile 32(M) x 64(N); double-buffered pipeline over 80 k-steps.
// All per-lane addressing is uniform-base + 32-bit element offsets (saddr+voffset form).
__global__ __launch_bounds__(256) void k_conv_gemm(
        const unsigned short* __restrict__ xpad, const unsigned short* __restrict__ wT,
        const float* __restrict__ conv_b, unsigned short* __restrict__ out) {
    const int lane = threadIdx.x & 31;
    const int wave = threadIdx.x >> 5;
    const int flat = blockIdx.x * 8 + wave;        // 2048 blocks * 8 waves = 16384 tiles
    const int tm = flat >> 3;                      // M/32 = 2048 tiles
    const int tn = flat & 7;                       // N/64 = 8 tiles
    const int m0 = tm << 5;
    const int n0 = tn << 6;
    const int b  = m0 >> 11;                       // /Tn  (32 | 2048 -> whole tile same batch)
    const int t0 = m0 & (Tn - 1);
    const int mrow = lane & 15;
    const int hi   = lane >> 4;

    const int aoff = (b * Tpad + t0 + mrow) * Cn + hi * 8;   // < 33.7M, fits int
    const int boff = (n0 + mrow) * Cn + hi * 8;              // < 1.4M with kk term

    f32x8 acc[2][4] = {};

    auto LOAD = [&](Frags& f, int q) {
        const int kk = q >> 4;
        const int k0 = (q & 15) << 5;
        const int a0 = aoff + kk * Cn + k0;
        f.a0.h[0] = *(const bf16x8*)(xpad + a0);
        f.a0.h[1] = *(const bf16x8*)(xpad + a0 + 16);
        f.a1.h[0] = *(const bf16x8*)(xpad + a0 + 16 * Cn);
        f.a1.h[1] = *(const bf16x8*)(xpad + a0 + 16 * Cn + 16);
        const int bo = boff + kk * (Cn * Cn) + k0;
        #pragma unroll
        for (int j = 0; j < 4; ++j) {
            const int bj = bo + j * 16 * Cn;
            f.bb[j].h[0] = *(const bf16x8*)(wT + bj);
            f.bb[j].h[1] = *(const bf16x8*)(wT + bj + 16);
        }
    };
    auto MMA = [&](Frags& f) {
        #pragma unroll
        for (int j = 0; j < 4; ++j) {
            acc[0][j] = wmma_bf16(f.a0.v, f.bb[j].v, acc[0][j]);
            acc[1][j] = wmma_bf16(f.a1.v, f.bb[j].v, acc[1][j]);
        }
    };

    Frags buf0, buf1;
    LOAD(buf0, 0);
    for (int q = 0; q < Kc * 16; q += 2) {         // 80 steps, 2 per iteration
        LOAD(buf1, q + 1);                          // prefetch while buf0 computes
        MMA(buf0);
        if (q + 2 < Kc * 16) LOAD(buf0, q + 2);     // prefetch while buf1 computes
        MMA(buf1);
    }

    // epilogue: +bias, store bf16.  C layout: vgpr r -> M = r + 8*hi, N = lane&15
    #pragma unroll
    for (int mi = 0; mi < 2; ++mi) {
        #pragma unroll
        for (int j = 0; j < 4; ++j) {
            const int col = n0 + j * 16 + mrow;
            const float bias = conv_b[col];
            const int obase = (m0 + mi * 16 + hi * 8) * Cn + col;   // < 33.6M, fits int
            #pragma unroll
            for (int r = 0; r < 8; ++r)
                out[obase + r * Cn] = f2bf(acc[mi][j][r] + bias);
        }
    }
}

// ---------------- kernel 4: dense bf16 WMMA GEMM + ReLU ----------------
__global__ __launch_bounds__(256) void k_dense_gemm(
        const unsigned short* __restrict__ A, const unsigned short* __restrict__ wT,
        const float* __restrict__ dense_b, unsigned short* __restrict__ out) {
    const int lane = threadIdx.x & 31;
    const int wave = threadIdx.x >> 5;
    const int flat = blockIdx.x * 8 + wave;
    const int tm = flat >> 3;
    const int tn = flat & 7;
    const int m0 = tm << 5;
    const int n0 = tn << 6;
    const int mrow = lane & 15;
    const int hi   = lane >> 4;

    const int aoff = (m0 + mrow) * Cn + hi * 8;    // < 33.6M, fits int
    const int boff = (n0 + mrow) * Cn + hi * 8;

    f32x8 acc[2][4] = {};

    auto LOAD = [&](Frags& f, int q) {
        const int k0 = q << 5;
        const int a0 = aoff + k0;
        f.a0.h[0] = *(const bf16x8*)(A + a0);
        f.a0.h[1] = *(const bf16x8*)(A + a0 + 16);
        f.a1.h[0] = *(const bf16x8*)(A + a0 + 16 * Cn);
        f.a1.h[1] = *(const bf16x8*)(A + a0 + 16 * Cn + 16);
        const int bo = boff + k0;
        #pragma unroll
        for (int j = 0; j < 4; ++j) {
            const int bj = bo + j * 16 * Cn;
            f.bb[j].h[0] = *(const bf16x8*)(wT + bj);
            f.bb[j].h[1] = *(const bf16x8*)(wT + bj + 16);
        }
    };
    auto MMA = [&](Frags& f) {
        #pragma unroll
        for (int j = 0; j < 4; ++j) {
            acc[0][j] = wmma_bf16(f.a0.v, f.bb[j].v, acc[0][j]);
            acc[1][j] = wmma_bf16(f.a1.v, f.bb[j].v, acc[1][j]);
        }
    };

    Frags buf0, buf1;
    LOAD(buf0, 0);
    for (int q = 0; q < 16; q += 2) {
        LOAD(buf1, q + 1);
        MMA(buf0);
        if (q + 2 < 16) LOAD(buf0, q + 2);
        MMA(buf1);
    }

    #pragma unroll
    for (int mi = 0; mi < 2; ++mi) {
        #pragma unroll
        for (int j = 0; j < 4; ++j) {
            const int col = n0 + j * 16 + mrow;
            const float bias = dense_b[col];
            const int obase = (m0 + mi * 16 + hi * 8) * Cn + col;
            #pragma unroll
            for (int r = 0; r < 8; ++r)
                out[obase + r * Cn] = f2bf(fmaxf(acc[mi][j][r] + bias, 0.0f));
        }
    }
}

// ---------------- kernel 5: wproj dot + sigmoid + mask + per-batch sum ----------------
__global__ __launch_bounds__(256) void k_wproj(
        const unsigned short* __restrict__ act, const float* __restrict__ wproj_w,
        const float* __restrict__ wproj_b, const int* __restrict__ pad_mask,
        float* __restrict__ weight, float* __restrict__ wsum) {
    const int lane = threadIdx.x & 31;
    const int wave = threadIdx.x >> 5;
    const int row = blockIdx.x * 8 + wave;         // one row per wave; < 65536
    const int abase = row * Cn + lane * 16;        // < 33.6M, fits int
    float s = 0.0f;
    #pragma unroll
    for (int i = 0; i < 16; ++i)
        s += bf2f(act[abase + i]) * wproj_w[lane * 16 + i];
    #pragma unroll
    for (int off = 16; off > 0; off >>= 1)
        s += __shfl_xor(s, off, 32);
    if (lane == 0) {
        float v = s + wproj_b[0];
        float sig = 1.0f / (1.0f + expf(-v));
        float np = 1.0f - (float)pad_mask[row];
        float w = sig * np;
        weight[row] = w;                        // org (unscaled) weight
        atomicAdd(&wsum[row >> 11], w);
    }
}

// ---------------- kernel 6: CIF scan + compaction (one block per batch) ----------------
__global__ __launch_bounds__(512) void k_scan(
        const float* __restrict__ x, const float* __restrict__ weight,
        const float* __restrict__ wsum, const int* __restrict__ tgt_len,
        const int* __restrict__ pad_mask,
        float* __restrict__ cif_out, float* __restrict__ mask_out,
        float* __restrict__ qty_out) {
    __shared__ float s_w[Tn];                      // staged weight row
    __shared__ int   s_fired[Tn];
    __shared__ float s_cacc[Tn];
    __shared__ float s_cemit[Tn];
    __shared__ int   s_slot[Tn];
    __shared__ int   s_n, s_ps;

    const int b = blockIdx.x;
    const int tid = threadIdx.x;

    if (tid == 0) s_ps = 0;
    __syncthreads();
    {   // stage weights to LDS + padding_start_id = sum(1 - pad_mask)
        int loc = 0;
        for (int t = tid; t < Tn; t += 512) {
            s_w[t] = weight[b * Tn + t];
            loc += 1 - pad_mask[b * Tn + t];
        }
        atomicAdd(&s_ps, loc);
    }
    __syncthreads();

    if (tid == 0) {
        const float scale = (float)tgt_len[b] / wsum[b];
        const int ps = s_ps;
        float accw = 0.0f;
        int cnt = 0;
        for (int t = 0; t < Tn; ++t) {
            const float w = s_w[t] * scale;
            const int fired = (accw + w) >= THRESH;
            const float remained = 1.0f - accw;
            s_fired[t] = fired;
            if (fired) {
                s_cemit[t] = remained;
                s_cacc[t]  = w - remained;
                s_slot[t]  = (t <= ps) ? cnt++ : -1;   // rows zeroed past padding aren't marks
                accw = w - remained;
            } else {
                s_cemit[t] = 0.0f;
                s_cacc[t]  = w;
                s_slot[t]  = -1;
                accw += w;
            }
        }
        s_n = cnt;
        qty_out[b] = wsum[b];                          // quantity_out = unscaled weight sum
    }
    __syncthreads();

    // vector replay: one channel per thread
    const int c = tid;                                 // Cn == 512 == blockDim.x
    const float* xb = x + (long long)b * Tn * Cn + c;
    float acc = 0.0f;
    for (int t = 0; t < Tn; ++t) {
        const float xv = xb[(long long)t * Cn];
        const float ca = s_cacc[t];
        if (s_fired[t]) {
            const int sl = s_slot[t];
            if (sl >= 0)
                cif_out[((long long)b * Tn + sl) * Cn + c] = acc + s_cemit[t] * xv;
            acc = ca * xv;
        } else {
            acc += ca * xv;
        }
    }
    // padding mask over compacted output
    for (int j = tid; j < Tn; j += 512)
        mask_out[b * Tn + j] = (j < s_n) ? 1.0f : 0.0f;
}

// ---------------- host: launch sequence ----------------
extern "C" void kernel_launch(void* const* d_in, const int* in_sizes, int n_in,
                              void* d_out, int out_size, void* d_ws, size_t ws_size,
                              hipStream_t stream) {
    (void)in_sizes; (void)n_in; (void)out_size; (void)ws_size;
    const float* x        = (const float*)d_in[0];
    const int*   pad_mask = (const int*)  d_in[1];
    const int*   tgt_len  = (const int*)  d_in[2];
    const float* conv_w   = (const float*)d_in[3];
    const float* conv_b   = (const float*)d_in[4];
    const float* dense_w  = (const float*)d_in[5];
    const float* dense_b  = (const float*)d_in[6];
    const float* wproj_w  = (const float*)d_in[7];
    const float* wproj_b  = (const float*)d_in[8];

    // workspace layout (bytes), all offsets 256-aligned
    char* ws = (char*)d_ws;
    const long long xpad_elems = (long long)Bn * Tpad * Cn;         // bf16
    unsigned short* xpad   = (unsigned short*)(ws);
    unsigned short* convo  = (unsigned short*)(ws + 67239936LL);            // after xpad
    unsigned short* act    = (unsigned short*)(ws + 67239936LL + 67108864LL);
    unsigned short* convWT = (unsigned short*)(ws + 201457664LL);
    unsigned short* denseWT= (unsigned short*)(ws + 204079104LL);
    float*          weight = (float*)        (ws + 204603392LL);
    float*          wsum   = (float*)        (ws + 204865536LL);

    float* out      = (float*)d_out;
    float* cif_out  = out;                                  // [B,T,C]
    float* mask_out = out + (long long)Bn * Tn * Cn;        // [B,T]
    float* qty_out  = mask_out + (long long)Bn * Tn;        // [B]
    const long long n_out = (long long)Bn * Tn * Cn + (long long)Bn * Tn + Bn;

    k_init<<<(unsigned)((n_out + 255) / 256), 256, 0, stream>>>(out, n_out, wsum);
    k_prep_x<<<(unsigned)((xpad_elems + 255) / 256), 256, 0, stream>>>(x, xpad);
    k_prep_w<<<(Kc * Cn * Cn + Cn * Cn + 255) / 256, 256, 0, stream>>>(conv_w, dense_w, convWT, denseWT);

    const int gemm_blocks = (Mtot / 32) * (Cn / 64) / 8;    // 2048
    k_conv_gemm <<<gemm_blocks, 256, 0, stream>>>(xpad, convWT, conv_b, convo);
    k_dense_gemm<<<gemm_blocks, 256, 0, stream>>>(convo, denseWT, dense_b, act);

    k_wproj<<<Mtot / 8, 256, 0, stream>>>(act, wproj_w, wproj_b, pad_mask, weight, wsum);
    k_scan <<<Bn, 512, 0, stream>>>(x, weight, wsum, tgt_len, pad_mask, cif_out, mask_out, qty_out);
}
